// GCNRecommender_85813446574383
// MI455X (gfx1250) — matrix operand order repaired and checked by reading, not compiled, not run on hardware
//
#include <hip/hip_runtime.h>
#include <hip/hip_bf16.h>

typedef __attribute__((ext_vector_type(8)))  _Float16 v8h;
typedef __attribute__((ext_vector_type(16))) _Float16 v16h;
typedef __attribute__((ext_vector_type(8)))  float    v8f;

#define NUM_USERS    100000
#define NUM_SERVICES 50000
#define EMB_DIM      32
#define CONTEXT      3
#define NUM_EDGES    1600000
#define BATCH        2048
#define FC_IN        128   // (CONTEXT+1)*EMB_DIM

// ---------------------------------------------------------------------------
// Degree / normalization
// ---------------------------------------------------------------------------
__global__ void init_deg_kernel(float* deg, int n) {
    int i = blockIdx.x * blockDim.x + threadIdx.x;
    if (i < n) deg[i] = 1.0f;  // self-loop
}

__global__ void count_deg_kernel(const int* __restrict__ dst, float* deg, int nE) {
    int e = blockIdx.x * blockDim.x + threadIdx.x;
    if (e < nE) atomicAdd(&deg[dst[e]], 1.0f);
}

__global__ void rsqrt_kernel(float* deg, int n) {
    int i = blockIdx.x * blockDim.x + threadIdx.x;
    if (i < n) deg[i] = rsqrtf(deg[i]);  // deg >= 1 always
}

// ---------------------------------------------------------------------------
// Small GEMM: h[n,32] = x[n,32] @ W[32,32]  (fp32 VALU, W cached in LDS)
// ---------------------------------------------------------------------------
__global__ void gemm32_kernel(const float* __restrict__ x, const float* __restrict__ W,
                              float* __restrict__ h, int n) {
    __shared__ float Ws[32 * 32];
    int t = threadIdx.x;  // 256 threads = 8 nodes
    for (int i = t; i < 1024; i += 256) Ws[i] = W[i];
    __syncthreads();
    int node = blockIdx.x * 8 + (t >> 5);
    int col  = t & 31;
    if (node >= n) return;
    const float* xr = x + (size_t)node * 32;
    float acc = 0.0f;
#pragma unroll
    for (int k = 0; k < 32; ++k) acc = fmaf(xr[k], Ws[k * 32 + col], acc);
    h[(size_t)node * 32 + col] = acc;
}

__global__ void zero_kernel(float* p, int n) {
    int i = blockIdx.x * blockDim.x + threadIdx.x;
    if (i < n) p[i] = 0.0f;
}

// ---------------------------------------------------------------------------
// Edge aggregation: g[dst] += h[src] * dis[src]*dis[dst]   (8 threads/edge, 4 dims each)
// ---------------------------------------------------------------------------
__global__ void scatter_kernel(const float* __restrict__ h, const int* __restrict__ src,
                               const int* __restrict__ dst, const float* __restrict__ dis,
                               float* g, int nE) {
    int tid = blockIdx.x * blockDim.x + threadIdx.x;  // nE*8 threads
    int e = tid >> 3;
    if (e >= nE) return;
    int q = (tid & 7) * 4;
    int s = src[e], d = dst[e];
    float norm = dis[s] * dis[d];
    const float4 v = *(const float4*)(h + (size_t)s * 32 + q);
    float* gp = g + (size_t)d * 32 + q;
    atomicAdd(gp + 0, v.x * norm);
    atomicAdd(gp + 1, v.y * norm);
    atomicAdd(gp + 2, v.z * norm);
    atomicAdd(gp + 3, v.w * norm);
}

// self-loop term + bias (+ optional relu):  g = g + h*dis^2 + b
__global__ void finish_kernel(const float* __restrict__ h, const float* __restrict__ dis,
                              const float* __restrict__ b, float* g, int n, int relu) {
    int tid = blockIdx.x * blockDim.x + threadIdx.x;  // n*32
    if (tid >= n * 32) return;
    int i = tid >> 5, d = tid & 31;
    float di = dis[i];
    float v = g[tid] + h[tid] * di * di + b[d];
    if (relu) v = fmaxf(v, 0.0f);
    g[tid] = v;
}

// ---------------------------------------------------------------------------
// Build x_f16[2048,128] = concat(user_emb[user_idx], g[context_idx].flat)
// ---------------------------------------------------------------------------
__global__ void build_x_kernel(const int* __restrict__ uidx, const int* __restrict__ cidx,
                               const float* __restrict__ uemb, const float* __restrict__ g,
                               _Float16* __restrict__ xh) {
    int tid = blockIdx.x * blockDim.x + threadIdx.x;  // BATCH*FC_IN
    if (tid >= BATCH * FC_IN) return;
    int bi = tid >> 7, d = tid & 127;
    float v;
    if (d < 32) {
        v = uemb[(size_t)uidx[bi] * 32 + d];
    } else {
        int c = (d - 32) >> 5, dd = d & 31;
        v = g[(size_t)cidx[bi * CONTEXT + c] * 32 + dd];
    }
    xh[tid] = (_Float16)v;
}

// ---------------------------------------------------------------------------
// Transpose + convert: Wt[n,k] (f16, k contiguous) from fc_W[k,n] (f32)
// ---------------------------------------------------------------------------
__global__ void transpose_w_kernel(const float* __restrict__ W, _Float16* __restrict__ Wt) {
    __shared__ _Float16 tile[32][33];
    int n0 = blockIdx.x * 32;   // N blocks: ceil(50000/32)
    int k0 = blockIdx.y * 32;   // K blocks: 128/32 = 4
    int tx = threadIdx.x, ty = threadIdx.y;  // (32,8)
#pragma unroll
    for (int i = 0; i < 32; i += 8) {
        int k = k0 + ty + i, n = n0 + tx;
        tile[ty + i][tx] = (n < NUM_SERVICES) ? (_Float16)W[(size_t)k * NUM_SERVICES + n]
                                              : (_Float16)0.0f;
    }
    __syncthreads();
#pragma unroll
    for (int i = 0; i < 32; i += 8) {
        int n = n0 + ty + i, k = k0 + tx;
        if (n < NUM_SERVICES) Wt[(size_t)n * FC_IN + k] = tile[tx][ty + i];
    }
}

// ---------------------------------------------------------------------------
// Big GEMM via WMMA f16->f32: out[2048,50000] = x[2048,128] @ W[128,50000] + b
// One wave (= one block of 32) computes a 16x80 output strip: 5 N-subtiles,
// K=128 -> 4 k-steps of 32 -> 20 straight-line WMMAs, no bounds checks
// (3125 N-tiles = 625 strips * 5, exact partition; all control flow scalar).
// Per k-step: load ALL 5 B fragments into distinct registers (one clause,
// immediate offsets off precomputed bases), then 5 back-to-back WMMAs with
// no register overlap -> no hazard NOPs, one loadcnt wait per k-step.
// A fragment (16x32 f16): lane half=L/16, m=L%16; chunks K=[32k+8h,+8) and
// K=[32k+16+8h,+8). B fragment (32x16 f16): lane column n=L%16, same K chunks as A
// (two 8-half contiguous chunks thanks to the Wt[n][k] layout).
// ---------------------------------------------------------------------------
__global__ void __launch_bounds__(32)
fc_gemm_kernel(const _Float16* __restrict__ xh, const _Float16* __restrict__ wt,
               const float* __restrict__ bias, float* __restrict__ out) {
    const int lane  = threadIdx.x;
    const int mt    = blockIdx.y;            // 0..127 (M tiles)
    const int nBase = blockIdx.x * 80;       // 0..624 strips, exact cover of 50000
    const int half  = lane >> 4;
    const int l16   = lane & 15;
    const int m     = mt * 16 + l16;

    v8f acc[5];
#pragma unroll
    for (int t = 0; t < 5; ++t) {
        float bv = bias[nBase + t * 16 + l16];
#pragma unroll
        for (int r = 0; r < 8; ++r) acc[t][r] = bv;
    }

    // base pointers with the lane-half displacement folded in; all k-step
    // displacements become compile-time immediate offsets on the loads
    const _Float16* aBase = xh + (size_t)m * FC_IN + half * 8;
    const _Float16* bBase[5];
#pragma unroll
    for (int t = 0; t < 5; ++t)
        bBase[t] = wt + (size_t)(nBase + t * 16 + l16) * FC_IN + half * 8;

#pragma unroll
    for (int kk = 0; kk < 4; ++kk) {
        // ---- A fragment: two contiguous 16B chunks at immediate offsets
        v8h a0 = *(const v8h*)(aBase + kk * 32);        // K = 32kk + 8h + [0,8)
        v8h a1 = *(const v8h*)(aBase + kk * 32 + 16);   // K = 32kk + 16 + 8h + [0,8)
        v16h a;
#pragma unroll
        for (int i = 0; i < 8; ++i) { a[i] = a0[i]; a[8 + i] = a1[i]; }

        // ---- all 5 B fragments first (distinct registers, one load clause)
        v16h b[5];
#pragma unroll
        for (int t = 0; t < 5; ++t) {
            v8h b0 = *(const v8h*)(bBase[t] + kk * 32);      // K = 32kk+8h+[0,8)
            v8h b1 = *(const v8h*)(bBase[t] + kk * 32 + 16); // K = 32kk+16+8h+[0,8)
#pragma unroll
            for (int i = 0; i < 8; ++i) { b[t][i] = b0[i]; b[t][8 + i] = b1[i]; }
        }

        // ---- 5 back-to-back WMMAs (no operand overlap between them)
#pragma unroll
        for (int t = 0; t < 5; ++t)
            acc[t] = __builtin_amdgcn_wmma_f32_16x16x32_f16(
                false, a, false, b[t], (short)0, acc[t], false, false);
    }

    // ---- store: C/D layout, VGPR r -> row (r + 8*half), col = l16.
    // Non-temporal: output (410 MB) is write-once, keep L2 for Wt/x.
#pragma unroll
    for (int t = 0; t < 5; ++t) {
        int n = nBase + t * 16 + l16;
        float* op = out + (size_t)(mt * 16 + half * 8) * NUM_SERVICES + n;
#pragma unroll
        for (int r = 0; r < 8; ++r)
            __builtin_nontemporal_store(acc[t][r], op + (size_t)r * NUM_SERVICES);
    }
}

// ---------------------------------------------------------------------------
extern "C" void kernel_launch(void* const* d_in, const int* in_sizes, int n_in,
                              void* d_out, int out_size, void* d_ws, size_t ws_size,
                              hipStream_t stream) {
    const int*   user_idx    = (const int*)  d_in[0];
    const int*   context_idx = (const int*)  d_in[1];
    const int*   edge_index  = (const int*)  d_in[2];
    const float* user_emb    = (const float*)d_in[3];
    const float* service_emb = (const float*)d_in[4];
    const float* W1          = (const float*)d_in[5];
    const float* b1          = (const float*)d_in[6];
    const float* W2          = (const float*)d_in[7];
    const float* b2          = (const float*)d_in[8];
    const float* fc_W        = (const float*)d_in[9];
    const float* fc_b        = (const float*)d_in[10];
    float* out = (float*)d_out;

    const int* src = edge_index;
    const int* dst = edge_index + NUM_EDGES;

    // workspace layout (256B aligned slices), ~26.3 MB total
    char* ws = (char*)d_ws;
    float*    dis = (float*)ws;                               // 200,192 B
    float*    h   = (float*)(ws + 200192);                    // 6.4 MB
    float*    g   = (float*)(ws + 200192 + 6400000);          // 6.4 MB
    _Float16* xh  = (_Float16*)(ws + 200192 + 12800000);      // 512 KB
    _Float16* wt  = (_Float16*)(ws + 200192 + 12800000 + 524288); // 12.8 MB

    const int NFEAT = NUM_SERVICES * EMB_DIM;  // 1,600,000

    // degree + D^-1/2
    init_deg_kernel<<<(NUM_SERVICES + 255) / 256, 256, 0, stream>>>(dis, NUM_SERVICES);
    count_deg_kernel<<<(NUM_EDGES + 255) / 256, 256, 0, stream>>>(dst, dis, NUM_EDGES);
    rsqrt_kernel<<<(NUM_SERVICES + 255) / 256, 256, 0, stream>>>(dis, NUM_SERVICES);

    // fc_W -> f16 transposed (independent of GCN chain; cheap)
    transpose_w_kernel<<<dim3((NUM_SERVICES + 31) / 32, FC_IN / 32), dim3(32, 8), 0, stream>>>(fc_W, wt);

    // ---- GCN layer 1 (relu)
    gemm32_kernel<<<NUM_SERVICES / 8, 256, 0, stream>>>(service_emb, W1, h, NUM_SERVICES);
    zero_kernel<<<(NFEAT + 255) / 256, 256, 0, stream>>>(g, NFEAT);
    scatter_kernel<<<(NUM_EDGES * 8) / 256, 256, 0, stream>>>(h, src, dst, dis, g, NUM_EDGES);
    finish_kernel<<<(NFEAT + 255) / 256, 256, 0, stream>>>(h, dis, b1, g, NUM_SERVICES, 1);

    // ---- GCN layer 2 (no relu); h reused, g overwritten in place after gemm
    gemm32_kernel<<<NUM_SERVICES / 8, 256, 0, stream>>>(g, W2, h, NUM_SERVICES);
    zero_kernel<<<(NFEAT + 255) / 256, 256, 0, stream>>>(g, NFEAT);
    scatter_kernel<<<(NUM_EDGES * 8) / 256, 256, 0, stream>>>(h, src, dst, dis, g, NUM_EDGES);
    finish_kernel<<<(NFEAT + 255) / 256, 256, 0, stream>>>(h, dis, b2, g, NUM_SERVICES, 0);

    // ---- build x (f16)
    build_x_kernel<<<(BATCH * FC_IN) / 256, 256, 0, stream>>>(user_idx, context_idx, user_emb, g, xh);

    // ---- big WMMA GEMM: 625 N-strips x 128 M-tiles, one wave per block
    fc_gemm_kernel<<<dim3(NUM_SERVICES / 80, BATCH / 16), 32, 0, stream>>>(xh, wt, fc_b, out);
}
